// STAGE2_29162827940001
// MI455X (gfx1250) — matrix operand order repaired and checked by reading, not compiled, not run on hardware
//
#include <hip/hip_runtime.h>
#include <hip/hip_bf16.h>

typedef __attribute__((ext_vector_type(16))) _Float16 v16h;
typedef __attribute__((ext_vector_type(8)))  float    v8f;
typedef unsigned int u32x4 __attribute__((ext_vector_type(4)));
typedef int          i32x4 __attribute__((ext_vector_type(4)));
typedef int          i32x8 __attribute__((ext_vector_type(8)));

#define DEV __device__ __forceinline__

#if defined(__has_builtin)
# if __has_builtin(__builtin_amdgcn_tensor_load_to_lds)
#  define HAVE_TDM 1
# endif
#endif
#ifndef HAVE_TDM
# define HAVE_TDM 0
#endif

// ---------------------------------------------------------------- constants
static constexpr long N0  = 60L * 60 * 36;      // 129600 voxels, full grid
static constexpr int  P0  = 254;                // ceil(N0/512)
static constexpr long NP0 = (long)P0 * 512;     // 130048
static constexpr long N1  = 30L * 30 * 18;      // 16200, half grid
static constexpr int  P1  = 32;
static constexpr long NP1 = (long)P1 * 512;     // 16384
static constexpr long N2  = 15L * 15 * 9;       // 2025

constexpr long alup(long x) { return (x + 255) & ~255L; }

// ---------------------------------------------------------------- workspace layout (bytes)
static constexpr long OFF_FEAT = 0;                                  // feat [N0,128] f32 / feat1 / up2
static constexpr long OFF_H16  = OFF_FEAT + alup(N0 * 128 * 4);      // h16 [NP0,128] f16 / conv t1
static constexpr long OFF_BIG  = OFF_H16 + alup(NP0 * 128 * 2);      // qkv|mid / vox0 f32+f16 / decoder
static constexpr long OFF_AT   = OFF_BIG + alup(N0 * 512 * 2);       // attnout [NP0,128] f16 / conv misc
// conv-stage suballoc inside AT arena
static constexpr long AT_T2   = OFF_AT;
static constexpr long AT_Y2H  = AT_T2  + alup(32L * 16200 * 4);
static constexpr long AT_T3   = AT_Y2H + alup(32L * 16200 * 2);
static constexpr long AT_XP   = AT_T3  + alup(128L * 16200 * 4);
static constexpr long AT_XPH  = AT_XP  + alup(128L * 16200 * 4);
static constexpr long AT_IDN  = AT_XPH + alup(128L * 16200 * 2);
// BIG arena aliases
static constexpr long BG_V0F = OFF_BIG;
static constexpr long BG_V0H = BG_V0F + alup(128L * N0 * 4);
static constexpr long BG_D2  = OFF_BIG;
static constexpr long BG_D1  = BG_D2 + alup(128L * N0 * 4);
static constexpr long BG_UP1 = BG_D1 + alup(128L * N1 * 4);
// dedicated regions
static constexpr long OFF_S1AF = OFF_AT + alup(NP0 * 128 * 2);
static constexpr long OFF_S1AH = OFF_S1AF + alup(128L * N1 * 4);
static constexpr long OFF_S1BF = OFF_S1AH + alup(128L * N1 * 2);
static constexpr long OFF_S1BH = OFF_S1BF + alup(128L * N1 * 4);
static constexpr long OFF_V1F  = OFF_S1BH + alup(128L * N1 * 2);
static constexpr long OFF_V1H  = OFF_V1F + alup(128L * N1 * 4);
static constexpr long OFF_S2AF = OFF_V1H + alup(128L * N1 * 2);
static constexpr long OFF_S2AH = OFF_S2AF + alup(256L * N2 * 4);
static constexpr long OFF_S2BF = OFF_S2AH + alup(256L * N2 * 2);
static constexpr long OFF_ORD0 = OFF_S2BF + alup(256L * N2 * 4);
static constexpr long OFF_RNK0 = OFF_ORD0 + alup(N0 * 4);
static constexpr long OFF_ORD1 = OFF_RNK0 + alup(N0 * 4);
static constexpr long OFF_RNK1 = OFF_ORD1 + alup(N1 * 4);
static constexpr long OFF_DENSE= OFF_RNK1 + alup(N1 * 4);
static constexpr long OFF_BSUM = OFF_DENSE + alup(262144L * 4);
static constexpr long OFF_SHIFT= OFF_BSUM + alup(1024L * 4);
static constexpr long OFF_COND = OFF_SHIFT + alup(64);
static constexpr long OFF_COND1= OFF_COND + alup(128L * 4);
static constexpr long OFF_MOD  = OFF_COND1 + alup(128L * 4);
static constexpr long OFF_MEAN = OFF_MOD + alup(768L * 4);
static constexpr long OFF_VAR  = OFF_MEAN + alup(256L * 4);
static constexpr long OFF_WBUF = OFF_VAR + alup(256L * 4);

// ---------------------------------------------------------------- WMMA helpers
DEV v8f wmma32(v16h a, v16h b, v8f c) {
    return __builtin_amdgcn_wmma_f32_16x16x32_f16(false, a, false, b, (short)0, c, false, false);
}

// A-fragment (16x32 f16), LDS layout [m][k] with row stride `stride` halves.
DEV v16h ldsA_frag(const _Float16* s, int stride, int m0, int k0) {
    int lane = threadIdx.x & 31;
    const _Float16* p = s + (long)(m0 + (lane & 15)) * stride + k0 + ((lane >> 4) << 3);
    v16h a;
#pragma unroll
    for (int j = 0; j < 8; ++j) a[j] = p[j];
#pragma unroll
    for (int j = 0; j < 8; ++j) a[8 + j] = p[16 + j];
    return a;
}

// A-fragment with only K=0..15 real (upper 16 K zero). LDS layout [m][0..15].
DEV v16h ldsA_frag_k16(const _Float16* s, int stride, int m0) {
    int lane = threadIdx.x & 31;
    const _Float16* p = s + (long)(m0 + (lane & 15)) * stride + ((lane >> 4) << 3);
    v16h a;
#pragma unroll
    for (int j = 0; j < 8; ++j) a[j] = p[j];
#pragma unroll
    for (int j = 0; j < 8; ++j) a[8 + j] = (_Float16)0.f;
    return a;
}

// B-fragment (32x16 f16): LDS layout [n][k], stride halves.
DEV v16h ldsB_frag(const _Float16* s, int stride, int n0, int k0) {
    int lane = threadIdx.x & 31;
    const _Float16* p = s + (long)(n0 + (lane & 15)) * stride + k0 + ((lane >> 4) << 3);
    v16h b;
#pragma unroll
    for (int j = 0; j < 8; ++j) b[j] = p[j];
#pragma unroll
    for (int j = 0; j < 8; ++j) b[8 + j] = p[16 + j];
    return b;
}

// B-fragment with only K=0..15 real: elements 8..15 (K 16..31) are zero.
DEV v16h ldsB_frag_k16(const _Float16* s, int stride, int n0) {
    int lane = threadIdx.x & 31;
    v16h b;
#pragma unroll
    for (int j = 0; j < 16; ++j) b[j] = (_Float16)0.f;
    {
        const _Float16* p = s + (long)(n0 + (lane & 15)) * stride + ((lane >> 4) << 3);
#pragma unroll
        for (int j = 0; j < 8; ++j) b[j] = p[j];
    }
    return b;
}

DEV float geluf(float x) {
    const float c = 0.7978845608028654f;
    float t = tanhf(c * (x + 0.044715f * x * x * x));
    return 0.5f * x * (1.f + t);
}

#if HAVE_TDM
// TDM: DMA a [rows<=128] x 32-half A tile (row stride lda halves) into LDS laid out
// as 40-half rows (pad_interval=16 dwords, pad_amount=4 dwords). Rows beyond
// tensor_dim1 are zero-filled by the TDM (replaces software M-tail masking).
DEV void tdm_load_tile_a(const _Float16* A, long lda, long M, long bm, long k0, unsigned lds_off) {
    unsigned long long ga = (unsigned long long)(const void*)(A + bm * lda + k0);
    long rem = M - bm;
    unsigned rows = (unsigned)(rem < 128 ? rem : 128);
    unsigned stride_dw = (unsigned)(lda >> 1);
    u32x4 g0;
    g0[0] = 1u;                                    // count=1 (valid descriptor)
    g0[1] = lds_off;                               // lds_addr
    g0[2] = (unsigned)ga;                          // global_addr[31:0]
    g0[3] = (unsigned)(ga >> 32) | (2u << 30);     // global_addr[56:32] | type=2
    i32x8 g1;
    g1[0] = (int)((2u << 16) | (1u << 20) | (3u << 22) | (3u << 25)); // 4B elems, pad 16dw->+4dw
    g1[1] = (int)(16u << 16);                      // tensor_dim0 = 16 dwords
    g1[2] = (int)((rows & 0xFFFFu) << 16);         // tensor_dim1 low
    g1[3] = (int)((rows >> 16) | (16u << 16));     // tensor_dim1 hi | tile_dim0 = 16 dwords
    g1[4] = 128;                                   // tile_dim1 = 128 rows, tile_dim2 = 0
    g1[5] = (int)stride_dw;                        // tensor_dim0_stride (dwords)
    g1[6] = 0;
    g1[7] = 0;
    i32x4 z4 = {0, 0, 0, 0};
#if __clang_major__ >= 23
    i32x8 z8 = {0, 0, 0, 0, 0, 0, 0, 0};
    __builtin_amdgcn_tensor_load_to_lds(g0, g1, z4, z4, z8, 0);
#else
    __builtin_amdgcn_tensor_load_to_lds(g0, g1, z4, z4, 0);
#endif
}
#endif

// ---------------------------------------------------------------- generic WMMA GEMM
// D[M,N] = act(A[M,K]f16 @ B[K,N]f16 + bias). K must be a multiple of 32.
__global__ __launch_bounds__(256) void gemm_f16(
    const _Float16* __restrict__ A, long lda,
    const _Float16* __restrict__ B, long ldb,
    const float* __restrict__ bias,
    long M, long N, long K, int act,
    float* __restrict__ Df, long ldf,
    _Float16* __restrict__ Dh, long ldh,
    float* __restrict__ R, long ldr,
    const int* __restrict__ rowmap, const float* __restrict__ gate)
{
    __shared__ __align__(16) _Float16 As[128 * 40];
    __shared__ __align__(16) _Float16 Bs[64 * 40];
    const long bm = (long)blockIdx.y * 128;
    const long bn = (long)blockIdx.x * 64;
    const int tid = threadIdx.x, lane = tid & 31, wave = tid >> 5;
    const int wm = (wave >> 1) * 32, wn = (wave & 1) * 32;

    v8f acc[2][2];
#pragma unroll
    for (int i = 0; i < 2; ++i)
#pragma unroll
        for (int j = 0; j < 2; ++j)
#pragma unroll
            for (int r = 0; r < 8; ++r) acc[i][j][r] = 0.f;

    for (long k0 = 0; k0 < K; k0 += 32) {
#if HAVE_TDM
        if (wave == 0) tdm_load_tile_a(A, lda, M, bm, k0, (unsigned)(unsigned long long)&As[0]);
#else
        {   // stage A: 128 rows x 32 halves
            int rrow = tid >> 1, cc = (tid & 1) * 16;
            long gr = bm + rrow;
            _Float16* dst = &As[rrow * 40 + cc];
            if (gr < M) {
                const _Float16* srcp = A + gr * lda + k0 + cc;
#pragma unroll
                for (int j = 0; j < 16; ++j) dst[j] = srcp[j];
            } else {
#pragma unroll
                for (int j = 0; j < 16; ++j) dst[j] = (_Float16)0.f;
            }
        }
#endif
        // stage B transposed into [n][k]: one b128 row-chunk per thread, scatter to LDS
        {
            int kk = tid >> 3, nn0 = (tid & 7) * 8;
            long gn = bn + nn0;
            const _Float16* bp = B + (k0 + kk) * ldb + gn;
            _Float16 tmp[8];
            if (gn + 8 <= N) {
#pragma unroll
                for (int j = 0; j < 8; ++j) tmp[j] = bp[j];
            } else {
#pragma unroll
                for (int j = 0; j < 8; ++j) tmp[j] = (gn + j < N) ? bp[j] : (_Float16)0.f;
            }
#pragma unroll
            for (int j = 0; j < 8; ++j) Bs[(nn0 + j) * 40 + kk] = tmp[j];
            if (k0 + 32 < K) __builtin_prefetch(bp + 32 * ldb, 0, 1);  // global_prefetch_b8
        }
#if HAVE_TDM
        if (wave == 0) __builtin_amdgcn_s_wait_tensorcnt(0);
#endif
        __syncthreads();
#pragma unroll
        for (int i = 0; i < 2; ++i) {
            v16h af = ldsA_frag(As, 40, wm + i * 16, 0);
#pragma unroll
            for (int j = 0; j < 2; ++j) {
                v16h bf = ldsB_frag(Bs, 40, wn + j * 16, 0);
                acc[i][j] = wmma32(af, bf, acc[i][j]);
            }
        }
        __syncthreads();
    }

    const int col16 = lane & 15, rof = 8 * (lane >> 4);
#pragma unroll
    for (int i = 0; i < 2; ++i) {
#pragma unroll
        for (int j = 0; j < 2; ++j) {
            long colg = bn + wn + j * 16 + col16;
#pragma unroll
            for (int r = 0; r < 8; ++r) {
                long rowg = bm + wm + i * 16 + rof + r;
                if (rowg < M && colg < N) {
                    float v = acc[i][j][r];
                    if (bias) v += bias[colg];
                    if (act == 1) v = fmaxf(v, 0.f);
                    else if (act == 2) v = geluf(v);
                    if (Df) Df[rowg * ldf + colg] = v;
                    if (Dh) Dh[rowg * ldh + colg] = (_Float16)v;
                    if (R) {
                        long rr = rowmap ? (long)rowmap[rowg] : rowg;
                        R[rr * ldr + colg] += gate[colg] * v;
                    }
                }
            }
        }
    }
}

// ---------------------------------------------------------------- flash attention, one (patch, head) per block
// qkv: [P*512, 384] f16 (q|k|v each 128 = 8 heads * 16). out: [P*512, 128] f16.
__global__ __launch_bounds__(256) void attn_kernel(const _Float16* __restrict__ qkv,
                                                   long nvalid,
                                                   _Float16* __restrict__ out)
{
    __shared__ __align__(16) _Float16 qs[512 * 16];
    __shared__ __align__(16) _Float16 ks[512 * 16];
    __shared__ __align__(16) _Float16 vts[16 * 512];
    __shared__ __align__(16) _Float16 pscr[8 * 16 * 32];

    const int patch = blockIdx.x >> 3;
    const int head  = blockIdx.x & 7;
    const int tid = threadIdx.x, lane = tid & 31, wave = tid >> 5;
    const long rowbase = (long)patch * 512;

    // Q/K rows are contiguous 16-half chunks: copy via async global->LDS DMA (b128).
    for (int e = tid; e < 1024; e += 256) {
        int i = e >> 1, h8 = (e & 1) * 8;
        long src = (rowbase + i) * 384 + head * 16 + h8;
        unsigned dq = (unsigned)(unsigned long long)&qs[i * 16 + h8];
        unsigned dk = (unsigned)(unsigned long long)&ks[i * 16 + h8];
        unsigned long long gq = (unsigned long long)(const void*)(qkv + src);
        unsigned long long gk = gq + 256;   // +128 halves
        asm volatile("global_load_async_to_lds_b128 %0, %1, off" :: "v"(dq), "v"(gq) : "memory");
        asm volatile("global_load_async_to_lds_b128 %0, %1, off" :: "v"(dk), "v"(gk) : "memory");
    }
    // V needs a transpose into [hd][key]: vector load + strided LDS scatter.
    for (int e = tid; e < 1024; e += 256) {
        int i = e >> 1, d8 = (e & 1) * 8;
        const _Float16* vp = qkv + (rowbase + i) * 384 + 256 + head * 16 + d8;
        _Float16 tv[8];
#pragma unroll
        for (int j = 0; j < 8; ++j) tv[j] = vp[j];
#pragma unroll
        for (int j = 0; j < 8; ++j) vts[(d8 + j) * 512 + i] = tv[j];
    }
    asm volatile("s_wait_asynccnt 0x0" ::: "memory");
    __syncthreads();

    const int m0 = wave * 64;
    v16h qf[4];
#pragma unroll
    for (int mt = 0; mt < 4; ++mt) qf[mt] = ldsA_frag_k16(qs, 16, m0 + mt * 16);

    v8f oacc[4];
    float mrun[4][8], lrun[4][8];
#pragma unroll
    for (int mt = 0; mt < 4; ++mt)
#pragma unroll
        for (int r = 0; r < 8; ++r) { oacc[mt][r] = 0.f; mrun[mt][r] = -3.0e38f; lrun[mt][r] = 0.f; }

    const long kvalid = nvalid - rowbase;
    _Float16* myp = &pscr[wave * 16 * 32];
    const int col = lane & 15, half = lane >> 4;

    for (int kb = 0; kb < 16; ++kb) {      // 32 keys per iteration
        v16h bf0 = ldsB_frag_k16(ks, 16, kb * 32);
        v16h bf1 = ldsB_frag_k16(ks, 16, kb * 32 + 16);
        v16h vf  = ldsB_frag(vts, 512, 0, kb * 32);
        const bool ok0 = (kb * 32 + col) < kvalid;
        const bool ok1 = (kb * 32 + 16 + col) < kvalid;
#pragma unroll
        for (int mt = 0; mt < 4; ++mt) {
            v8f z;
#pragma unroll
            for (int r = 0; r < 8; ++r) z[r] = 0.f;
            v8f s0 = wmma32(qf[mt], bf0, z);
            v8f s1 = wmma32(qf[mt], bf1, z);
#pragma unroll
            for (int r = 0; r < 8; ++r) {
                float a = ok0 ? s0[r] * 0.25f : -1e9f;   // hd^-0.5 = 0.25
                float b = ok1 ? s1[r] * 0.25f : -1e9f;
                float mx = fmaxf(a, b);
#pragma unroll
                for (int d = 8; d; d >>= 1) mx = fmaxf(mx, __shfl_xor(mx, d, 32));
                float mnew = fmaxf(mrun[mt][r], mx);
                float p0 = __expf(a - mnew);
                float p1 = __expf(b - mnew);
                float ps = p0 + p1;
#pragma unroll
                for (int d = 8; d; d >>= 1) ps += __shfl_xor(ps, d, 32);
                float alpha = __expf(mrun[mt][r] - mnew);
                lrun[mt][r] = lrun[mt][r] * alpha + ps;
                mrun[mt][r] = mnew;
                oacc[mt][r] *= alpha;
                myp[(r + 8 * half) * 32 + col]      = (_Float16)p0;
                myp[(r + 8 * half) * 32 + 16 + col] = (_Float16)p1;
            }
            asm volatile("s_wait_dscnt 0x0" ::: "memory");   // P tile visible before A-frag read
            v16h pf = ldsA_frag(myp, 32, 0, 0);
            oacc[mt] = wmma32(pf, vf, oacc[mt]);
        }
    }

#pragma unroll
    for (int mt = 0; mt < 4; ++mt)
#pragma unroll
        for (int r = 0; r < 8; ++r) {
            int row = m0 + mt * 16 + r + 8 * half;
            float l = lrun[mt][r];
            float o = (l > 0.f) ? oacc[mt][r] / l : 0.f;
            out[(rowbase + row) * 128 + head * 16 + col] = (_Float16)o;
        }
}

// ---------------------------------------------------------------- small kernels
__global__ void argmax16(const float* __restrict__ logits, int* __restrict__ shifts) {
    if (threadIdx.x != 0) return;
    int r = blockIdx.x;
    const float* p = logits + r * 32;
    int best = 0; float bv = p[0];
    for (int i = 1; i < 32; ++i) if (p[i] > bv) { bv = p[i]; best = i; }
    shifts[r] = best * 16;
}

DEV int spread3(int v) { int o = 0; for (int i = 0; i < 6; ++i) o |= ((v >> i) & 1) << (3 * i); return o; }

__global__ void fill_i32_k(int* p, long n, int v) {
    long t = blockIdx.x * 256L + threadIdx.x; if (t < n) p[t] = v;
}
__global__ void morton_mark(int X, int Y, int Z, int* __restrict__ dense) {
    long t = blockIdx.x * 256L + threadIdx.x;
    long n = (long)X * Y * Z; if (t >= n) return;
    int z = t % Z; long r = t / Z; int y = r % Y; int x = r / Y;
    int code = (spread3(x) << 2) | (spread3(y) << 1) | spread3(z);
    dense[code] = (int)t;
}
__global__ void count_blocks(const int* __restrict__ dense, int* __restrict__ bsum) {
    __shared__ int sh[256];
    int t = threadIdx.x;
    sh[t] = dense[blockIdx.x * 256 + t] >= 0;
    __syncthreads();
    for (int s = 128; s; s >>= 1) { if (t < s) sh[t] += sh[t + s]; __syncthreads(); }
    if (!t) bsum[blockIdx.x] = sh[0];
}
__global__ void scan_serial(int* bsum, int nb) {
    if (threadIdx.x || blockIdx.x) return;
    int acc = 0;
    for (int i = 0; i < nb; ++i) { int v = bsum[i]; bsum[i] = acc; acc += v; }
}
__global__ void rank_kernel(const int* __restrict__ dense, const int* __restrict__ bsum, int* __restrict__ rank) {
    __shared__ int sh[256];
    int t = threadIdx.x;
    int v = dense[blockIdx.x * 256 + t];
    int occ = v >= 0;
    sh[t] = occ; __syncthreads();
    for (int s = 1; s < 256; s <<= 1) {
        int add = (t >= s) ? sh[t - s] : 0;
        __syncthreads();
        sh[t] += add;
        __syncthreads();
    }
    if (occ) rank[v] = bsum[blockIdx.x] + sh[t] - 1;
}
__global__ void build_order(const int* __restrict__ rank, long n, const int* __restrict__ shifts,
                            int which, int* __restrict__ order) {
    long v = blockIdx.x * 256L + threadIdx.x; if (v >= n) return;
    long pos = rank[v] + shifts[which]; if (pos >= n) pos -= n;
    order[pos] = (int)v;
}

__global__ void gather_feat(const float* __restrict__ f2d, const int* __restrict__ depth,
                            float* __restrict__ feat) {
    long t = blockIdx.x * 256L + threadIdx.x; if (t >= N0 * 128) return;
    long v = t >> 7; int c = t & 127;
    feat[t] = f2d[c * 4800 + depth[v]];
}

// per-channel mean / (mean,var). elem(c,i) = x[c*cstride + i*istride]
__global__ void chan_reduce(const float* __restrict__ x, long n, long cstride, long istride,
                            float* __restrict__ mean, float* __restrict__ var) {
    __shared__ float ss[256], sq[256];
    int c = blockIdx.x;
    const float* p = x + (long)c * cstride;
    float s = 0.f, q = 0.f;
    for (long i = threadIdx.x; i < n; i += 256) { float v = p[i * istride]; s += v; q += v * v; }
    ss[threadIdx.x] = s; sq[threadIdx.x] = q; __syncthreads();
    for (int d = 128; d; d >>= 1) {
        if (threadIdx.x < d) { ss[threadIdx.x] += ss[threadIdx.x + d]; sq[threadIdx.x] += sq[threadIdx.x + d]; }
        __syncthreads();
    }
    if (!threadIdx.x) {
        float m = ss[0] / (float)n;
        mean[c] = m;
        if (var) var[c] = sq[0] / (float)n - m * m;
    }
}

__global__ void ada_mod(const float* __restrict__ cond, const float* __restrict__ Wada,
                        const float* __restrict__ bada, float* __restrict__ mod) {
    int o = blockIdx.x * 256 + threadIdx.x; if (o >= 768) return;
    float acc = bada[o];
    for (int c = 0; c < 128; ++c) {
        float s = cond[c];
        s = s / (1.f + __expf(-s));
        acc += s * Wada[c * 768 + o];
    }
    mod[o] = acc;
}

// h[i,:] = LN(feat[order?order[i]:i, :]) * (1+sc) + sh ; rows >= Nrows are zero-padded.
__global__ void ln_mod_kernel(const float* __restrict__ feat, long Nrows, long Npad,
                              const float* __restrict__ shv, const float* __restrict__ scv,
                              const int* __restrict__ order, _Float16* __restrict__ out) {
    long row = (blockIdx.x * 256L + threadIdx.x) >> 5;
    int lane = threadIdx.x & 31;
    if (row >= Npad) return;
    long o = row * 128;
    if (row >= Nrows) {
        for (int j = 0; j < 4; ++j) out[o + lane * 4 + j] = (_Float16)0.f;
        return;
    }
    long src = order ? (long)order[row] : row;
    const float* x = feat + src * 128;
    float v[4];
#pragma unroll
    for (int j = 0; j < 4; ++j) v[j] = x[lane * 4 + j];
    float s = v[0] + v[1] + v[2] + v[3];
    float q = v[0] * v[0] + v[1] * v[1] + v[2] * v[2] + v[3] * v[3];
#pragma unroll
    for (int d = 16; d; d >>= 1) { s += __shfl_xor(s, d, 32); q += __shfl_xor(q, d, 32); }
    float m = s * (1.f / 128.f);
    float inv = rsqrtf(q * (1.f / 128.f) - m * m + 1e-5f);
#pragma unroll
    for (int j = 0; j < 4; ++j) {
        int c = lane * 4 + j;
        out[o + c] = (_Float16)((v[j] - m) * inv * (1.f + scv[c]) + shv[c]);
    }
}

__global__ void cvt_f16_k(const float* __restrict__ x, _Float16* __restrict__ y, long n) {
    long t = blockIdx.x * 256L + threadIdx.x; if (t < n) y[t] = (_Float16)x[t];
}
__global__ void feat_to_vox(const float* __restrict__ feat, long n, float* __restrict__ vox,
                            _Float16* __restrict__ voxh) {
    long t = blockIdx.x * 256L + threadIdx.x; if (t >= n * 128) return;
    long v = t >> 7; int c = t & 127;
    float val = feat[t];
    long o = (long)c * n + v;
    vox[o] = val;
    if (voxh) voxh[o] = (_Float16)val;
}
__global__ void vox_to_feat(const float* __restrict__ vox, long n, float* __restrict__ feat) {
    long t = blockIdx.x * 256L + threadIdx.x; if (t >= n * 128) return;
    long v = t >> 7; int c = t & 127;
    feat[t] = vox[(long)c * n + v];
}

// channel-major batchnorm apply with optional residual/relu/f16 mirror. grid (ceil(n/256), C)
__global__ void bn_apply(const float* __restrict__ x, long n,
                         const float* __restrict__ mean, const float* __restrict__ var,
                         const float* __restrict__ g, const float* __restrict__ b,
                         int relu, int resid_after, const float* __restrict__ resid,
                         float* __restrict__ outf, _Float16* __restrict__ outh) {
    int c = blockIdx.y;
    long i = blockIdx.x * 256L + threadIdx.x; if (i >= n) return;
    long t = (long)c * n + i;
    float v = g[c] * (x[t] - mean[c]) * rsqrtf(var[c] + 1e-5f) + b[c];
    if (resid && !resid_after) v += resid[t];
    if (relu) v = fmaxf(v, 0.f);
    if (resid && resid_after) v += resid[t];
    if (outf) outf[t] = v;
    if (outh) outh[t] = (_Float16)v;
}

__global__ void conv3x3k(const float* __restrict__ x, int Ci, int Di, int Hi, int Wi,
                         const float* __restrict__ w, int Co, int stride,
                         float* __restrict__ y, int Do, int Ho, int Wo) {
    long t = blockIdx.x * 256L + threadIdx.x;
    long tot = (long)Co * Do * Ho * Wo; if (t >= tot) return;
    int ow = t % Wo; long r = t / Wo; int oh = r % Ho; r /= Ho; int od = r % Do; int co = (int)(r / Do);
    long plane = (long)Di * Hi * Wi;
    float acc = 0.f;
    for (int kd = 0; kd < 3; ++kd) {
        int id = od * stride + kd - 1; if (id < 0 || id >= Di) continue;
        for (int kh = 0; kh < 3; ++kh) {
            int ih = oh * stride + kh - 1; if (ih < 0 || ih >= Hi) continue;
            for (int kw = 0; kw < 3; ++kw) {
                int iw = ow * stride + kw - 1; if (iw < 0 || iw >= Wi) continue;
                long xoff = ((long)id * Hi + ih) * Wi + iw;
                const float* wp = w + (long)co * Ci * 27 + kd * 9 + kh * 3 + kw;
                for (int ci = 0; ci < Ci; ++ci) acc += x[ci * plane + xoff] * wp[ci * 27];
            }
        }
    }
    y[t] = acc;
}

__global__ void avgpool2k(const float* __restrict__ x, int Cc, int Di, int Hi, int Wi,
                          float* __restrict__ y) {
    int Do = Di / 2, Ho = Hi / 2, Wo = Wi / 2;
    long t = blockIdx.x * 256L + threadIdx.x;
    long tot = (long)Cc * Do * Ho * Wo; if (t >= tot) return;
    int ow = t % Wo; long r = t / Wo; int oh = r % Ho; r /= Ho; int od = r % Do; int c = (int)(r / Do);
    const float* xc = x + (long)c * Di * Hi * Wi;
    float s = 0.f;
    for (int a = 0; a < 2; ++a)
        for (int b = 0; b < 2; ++b)
            for (int d = 0; d < 2; ++d)
                s += xc[((long)(od * 2 + a) * Hi + (oh * 2 + b)) * Wi + (ow * 2 + d)];
    y[t] = s * 0.125f;
}

// lhs-dilated conv == ConvTranspose3d(k=3,s=2,p=1,op=1). w: [Co][Ci][27]
__global__ void deconv3k(const float* __restrict__ x, int Ci, int Di, int Hi, int Wi,
                         const float* __restrict__ w, const float* __restrict__ bias, int Co,
                         float* __restrict__ y) {
    int Do = 2 * Di, Ho = 2 * Hi, Wo = 2 * Wi;
    long t = blockIdx.x * 256L + threadIdx.x;
    long tot = (long)Co * Do * Ho * Wo; if (t >= tot) return;
    int ow = t % Wo; long r = t / Wo; int oh = r % Ho; r /= Ho; int od = r % Do; int co = (int)(r / Do);
    long plane = (long)Di * Hi * Wi;
    float acc = bias[co];
    for (int kd = 0; kd < 3; ++kd) {
        int dz = od + kd - 1; if (dz < 0 || (dz & 1) || (dz >> 1) >= Di) continue;
        int id = dz >> 1;
        for (int kh = 0; kh < 3; ++kh) {
            int dy = oh + kh - 1; if (dy < 0 || (dy & 1) || (dy >> 1) >= Hi) continue;
            int ih = dy >> 1;
            for (int kw = 0; kw < 3; ++kw) {
                int dx = ow + kw - 1; if (dx < 0 || (dx & 1) || (dx >> 1) >= Wi) continue;
                int iw = dx >> 1;
                long xoff = ((long)id * Hi + ih) * Wi + iw;
                const float* wp = w + (long)co * Ci * 27 + kd * 9 + kh * 3 + kw;
                for (int ci = 0; ci < Ci; ++ci) acc += x[ci * plane + xoff] * wp[ci * 27];
            }
        }
    }
    y[t] = acc;
}

__global__ void trilinear_add(const float* __restrict__ x, int Cc, int Di, int Hi, int Wi,
                              float* __restrict__ y, int Do, int Ho, int Wo) {
    long t = blockIdx.x * 256L + threadIdx.x;
    long tot = (long)Cc * Do * Ho * Wo; if (t >= tot) return;
    int ow = t % Wo; long r = t / Wo; int oh = r % Ho; r /= Ho; int od = r % Do; int c = (int)(r / Do);
    float pd = od * (float)(Di - 1) / (float)(Do - 1);
    float ph = oh * (float)(Hi - 1) / (float)(Ho - 1);
    float pw = ow * (float)(Wi - 1) / (float)(Wo - 1);
    int d0 = (int)pd, h0 = (int)ph, w0 = (int)pw;
    int d1 = d0 + 1 < Di ? d0 + 1 : Di - 1;
    int h1 = h0 + 1 < Hi ? h0 + 1 : Hi - 1;
    int w1 = w0 + 1 < Wi ? w0 + 1 : Wi - 1;
    float fd = pd - d0, fh = ph - h0, fw = pw - w0;
    const float* xc = x + (long)c * Di * Hi * Wi;
#define AT3(a, b, cc) xc[((long)(a) * Hi + (b)) * Wi + (cc)]
    float v =
        AT3(d0, h0, w0) * (1 - fd) * (1 - fh) * (1 - fw) + AT3(d0, h0, w1) * (1 - fd) * (1 - fh) * fw +
        AT3(d0, h1, w0) * (1 - fd) * fh * (1 - fw) + AT3(d0, h1, w1) * (1 - fd) * fh * fw +
        AT3(d1, h0, w0) * fd * (1 - fh) * (1 - fw) + AT3(d1, h0, w1) * fd * (1 - fh) * fw +
        AT3(d1, h1, w0) * fd * fh * (1 - fw) + AT3(d1, h1, w1) * fd * fh * fw;
#undef AT3
    y[t] += v;
}

__global__ void conv_out_k(const float* __restrict__ x, const float* __restrict__ w,
                           const float* __restrict__ b, float* __restrict__ out, long n) {
    long t = blockIdx.x * 256L + threadIdx.x;
    long tot = 12 * n; if (t >= tot) return;
    int cls = (int)(t / n); long i = t % n;
    float acc = b[cls];
    const float* wr = w + cls * 128;
    for (int c = 0; c < 128; ++c) acc += x[(long)c * n + i] * wr[c];
    out[t] = acc;
}

// ---------------------------------------------------------------- host helpers
static inline unsigned gr1(long n) { return (unsigned)((n + 255) / 256); }

static void launch_gemm(hipStream_t st, const _Float16* A, long lda, const _Float16* B, long ldb,
                        const float* bias, long M, long N, long K, int act,
                        float* Df, long ldf, _Float16* Dh, long ldh,
                        float* R, long ldr, const int* rowmap, const float* gate) {
    dim3 grid((unsigned)((N + 63) / 64), (unsigned)((M + 127) / 128));
    gemm_f16<<<grid, 256, 0, st>>>(A, lda, B, ldb, bias, M, N, K, act, Df, ldf, Dh, ldh, R, ldr, rowmap, gate);
}

static void run_serialize(hipStream_t st, char* ws, int X, int Y, int Z, int* rank) {
    int* dense = (int*)(ws + OFF_DENSE);
    int* bsum  = (int*)(ws + OFF_BSUM);
    long n = (long)X * Y * Z;
    fill_i32_k<<<1024, 256, 0, st>>>(dense, 262144, -1);
    morton_mark<<<gr1(n), 256, 0, st>>>(X, Y, Z, dense);
    count_blocks<<<1024, 256, 0, st>>>(dense, bsum);
    scan_serial<<<1, 1, 0, st>>>(bsum, 1024);
    rank_kernel<<<1024, 256, 0, st>>>(dense, bsum, rank);
}

static void run_block(hipStream_t st, char* ws, float* feat, long Nr, long Npad, int P,
                      const int* order, const float* cond,
                      const float* Wada, const float* bada,
                      const float* Wqkv, const float* bqkv,
                      const float* Wp, const float* bp,
                      const float* W1, const float* b1,
                      const float* W2, const float* b2) {
    float* mod = (float*)(ws + OFF_MOD);
    _Float16* h16  = (_Float16*)(ws + OFF_H16);
    _Float16* qkv  = (_Float16*)(ws + OFF_BIG);
    _Float16* at   = (_Float16*)(ws + OFF_AT);
    _Float16* wbuf = (_Float16*)(ws + OFF_WBUF);

    ada_mod<<<3, 256, 0, st>>>(cond, Wada, bada, mod);
    // ---- attention branch: h = LN(feat)*(1+sc1)+s1, serialized + padded
    ln_mod_kernel<<<gr1(Npad * 32), 256, 0, st>>>(feat, Nr, Npad, mod + 0, mod + 128, order, h16);
    cvt_f16_k<<<gr1(128 * 384), 256, 0, st>>>(Wqkv, wbuf, 128 * 384);
    launch_gemm(st, h16, 128, wbuf, 384, bqkv, Npad, 384, 128, 0,
                nullptr, 0, qkv, 384, nullptr, 0, nullptr, nullptr);
    attn_kernel<<<dim3((unsigned)(P * 8)), 256, 0, st>>>(qkv, Nr, at);
    cvt_f16_k<<<gr1(128 * 128), 256, 0, st>>>(Wp, wbuf, 128 * 128);
    launch_gemm(st, at, 128, wbuf, 128, bp, Nr, 128, 128, 0,
                nullptr, 0, nullptr, 0, feat, 128, order, mod + 256);   // feat[order[i]] += g1*proj
    // ---- MLP branch
    ln_mod_kernel<<<gr1(Nr * 32), 256, 0, st>>>(feat, Nr, Nr, mod + 384, mod + 512, nullptr, h16);
    _Float16* mid = qkv;   // alias BIG arena
    cvt_f16_k<<<gr1(128 * 512), 256, 0, st>>>(W1, wbuf, 128 * 512);
    launch_gemm(st, h16, 128, wbuf, 512, b1, Nr, 512, 128, 2 /*gelu*/,
                nullptr, 0, mid, 512, nullptr, 0, nullptr, nullptr);
    cvt_f16_k<<<gr1(512 * 128), 256, 0, st>>>(W2, wbuf, 512 * 128);
    launch_gemm(st, mid, 512, wbuf, 128, b2, Nr, 128, 512, 0,
                nullptr, 0, nullptr, 0, feat, 128, nullptr, mod + 640); // feat += g2*mlp
}

static void run_bottleneck(hipStream_t st, char* ws,
                           const float* xf, const _Float16* xh,
                           int D, int H, int W, int stride,
                           int Cin, int planes, int Cout,
                           const float* w1, const float* g1, const float* b1,
                           const float* w2, const float* g2, const float* b2,
                           const float* w3, const float* g3, const float* b3,
                           const float* wd, const float* gd, const float* bd,
                           float* outf, _Float16* outh) {
    long nin = (long)D * H * W;
    int Do = (stride == 2) ? D / 2 : D, Ho = (stride == 2) ? H / 2 : H, Wo = (stride == 2) ? W / 2 : W;
    long nout = (long)Do * Ho * Wo;
    _Float16* wbuf = (_Float16*)(ws + OFF_WBUF);
    float* t1 = (float*)(ws + OFF_H16);
    float* t2 = (float*)(ws + AT_T2);
    _Float16* y2h = (_Float16*)(ws + AT_Y2H);
    float* t3 = (float*)(ws + AT_T3);
    float* xp = (float*)(ws + AT_XP);
    _Float16* xph = (_Float16*)(ws + AT_XPH);
    float* idn = (float*)(ws + AT_IDN);
    float* mean = (float*)(ws + OFF_MEAN);
    float* var  = (float*)(ws + OFF_VAR);

    // conv1x1 -> bn -> relu
    cvt_f16_k<<<gr1((long)planes * Cin), 256, 0, st>>>(w1, wbuf, (long)planes * Cin);
    launch_gemm(st, wbuf, Cin, xh, nin, nullptr, planes, nin, Cin, 0,
                t1, nin, nullptr, 0, nullptr, 0, nullptr, nullptr);
    chan_reduce<<<planes, 256, 0, st>>>(t1, nin, nin, 1, mean, var);
    bn_apply<<<dim3(gr1(nin), planes), 256, 0, st>>>(t1, nin, mean, var, g1, b1, 1, 0, nullptr, t1, nullptr);
    // conv3x3 -> bn -> relu (+f16)
    conv3x3k<<<gr1((long)planes * nout), 256, 0, st>>>(t1, planes, D, H, W, w2, planes, stride, t2, Do, Ho, Wo);
    chan_reduce<<<planes, 256, 0, st>>>(t2, nout, nout, 1, mean, var);
    bn_apply<<<dim3(gr1(nout), planes), 256, 0, st>>>(t2, nout, mean, var, g2, b2, 1, 0, nullptr, t2, y2h);
    // conv1x1 (no act yet)
    cvt_f16_k<<<gr1((long)Cout * planes), 256, 0, st>>>(w3, wbuf, (long)Cout * planes);
    launch_gemm(st, wbuf, planes, y2h, nout, nullptr, Cout, nout, planes, 0,
                t3, nout, nullptr, 0, nullptr, 0, nullptr, nullptr);
    // identity path
    const float* idnp;
    if (wd) {
        avgpool2k<<<gr1((long)Cin * nout), 256, 0, st>>>(xf, Cin, D, H, W, xp);
        cvt_f16_k<<<gr1((long)Cin * nout), 256, 0, st>>>(xp, xph, (long)Cin * nout);
        cvt_f16_k<<<gr1((long)Cout * Cin), 256, 0, st>>>(wd, wbuf, (long)Cout * Cin);
        launch_gemm(st, wbuf, Cin, xph, nout, nullptr, Cout, nout, Cin, 0,
                    idn, nout, nullptr, 0, nullptr, 0, nullptr, nullptr);
        chan_reduce<<<Cout, 256, 0, st>>>(idn, nout, nout, 1, mean, var);
        bn_apply<<<dim3(gr1(nout), Cout), 256, 0, st>>>(idn, nout, mean, var, gd, bd, 0, 0, nullptr, idn, nullptr);
        idnp = idn;
    } else {
        idnp = xf;
    }
    // out = relu(bn(t3) + idn)
    chan_reduce<<<Cout, 256, 0, st>>>(t3, nout, nout, 1, mean, var);
    bn_apply<<<dim3(gr1(nout), Cout), 256, 0, st>>>(t3, nout, mean, var, g3, b3, 1, 0, idnp, outf, outh);
}

// ---------------------------------------------------------------- entry
extern "C" void kernel_launch(void* const* d_in, const int* in_sizes, int n_in,
                              void* d_out, int out_size, void* d_ws, size_t ws_size,
                              hipStream_t stream) {
    (void)in_sizes; (void)n_in; (void)out_size; (void)ws_size;
    char* ws = (char*)d_ws;
    auto F = [&](int i) { return (const float*)d_in[i]; };

    const float* f2d   = F(0);
    const int*   depth = (const int*)d_in[1];
    const float* slog  = F(85);

    int* shifts = (int*)(ws + OFF_SHIFT);
    int* rank0  = (int*)(ws + OFF_RNK0);
    int* rank1  = (int*)(ws + OFF_RNK1);
    int* order0 = (int*)(ws + OFF_ORD0);
    int* order1 = (int*)(ws + OFF_ORD1);

    argmax16<<<3, 32, 0, stream>>>(slog, shifts);
    run_serialize(stream, ws, 60, 60, 36, rank0);
    run_serialize(stream, ws, 30, 30, 18, rank1);

    // 2D -> 3D projection + cond
    float* feat = (float*)(ws + OFF_FEAT);
    gather_feat<<<gr1(N0 * 128), 256, 0, stream>>>(f2d, depth, feat);
    float* cond  = (float*)(ws + OFF_COND);
    float* cond1 = (float*)(ws + OFF_COND1);
    chan_reduce<<<128, 256, 0, stream>>>(feat, N0, 1, 128, cond, nullptr);

    // transformer blocks b0, b1 (params base 3, 10 tensors per block)
    for (int b = 0; b < 2; ++b) {
        int base = 3 + b * 10;
        build_order<<<gr1(N0), 256, 0, stream>>>(rank0, N0, shifts, b, order0);
        run_block(stream, ws, feat, N0, NP0, P0, order0, cond,
                  F(base + 0), F(base + 1), F(base + 2), F(base + 3), F(base + 4),
                  F(base + 5), F(base + 6), F(base + 7), F(base + 8), F(base + 9));
    }

    // to channel-major voxels
    float* v0f = (float*)(ws + BG_V0F);
    _Float16* v0h = (_Float16*)(ws + BG_V0H);
    feat_to_vox<<<gr1(N0 * 128), 256, 0, stream>>>(feat, N0, v0f, v0h);

    // sem1 = bottleneck(bottleneck(vox, k0, stride2), k1, stride1)
    float* s1af = (float*)(ws + OFF_S1AF);
    _Float16* s1ah = (_Float16*)(ws + OFF_S1AH);
    run_bottleneck(stream, ws, v0f, v0h, 60, 60, 36, 2, 128, 32, 128,
                   F(33), F(34), F(35), F(36), F(37), F(38), F(39), F(40), F(41),
                   F(42), F(43), F(44), s1af, s1ah);
    float* sem1f = (float*)(ws + OFF_S1BF);
    _Float16* sem1h = (_Float16*)(ws + OFF_S1BH);
    run_bottleneck(stream, ws, s1af, s1ah, 30, 30, 18, 1, 128, 32, 128,
                   F(45), F(46), F(47), F(48), F(49), F(50), F(51), F(52), F(53),
                   nullptr, nullptr, nullptr, sem1f, sem1h);

    // transformer block b2 at half resolution
    float* feat1 = feat;   // reuse arena
    vox_to_feat<<<gr1(N1 * 128), 256, 0, stream>>>(sem1f, N1, feat1);
    chan_reduce<<<128, 256, 0, stream>>>(sem1f, N1, N1, 1, cond1, nullptr);
    build_order<<<gr1(N1), 256, 0, stream>>>(rank1, N1, shifts, 2, order1);
    {
        int base = 3 + 2 * 10;
        run_block(stream, ws, feat1, N1, NP1, P1, order1, cond1,
                  F(base + 0), F(base + 1), F(base + 2), F(base + 3), F(base + 4),
                  F(base + 5), F(base + 6), F(base + 7), F(base + 8), F(base + 9));
    }
    float* v1f = (float*)(ws + OFF_V1F);
    _Float16* v1h = (_Float16*)(ws + OFF_V1H);
    feat_to_vox<<<gr1(N1 * 128), 256, 0, stream>>>(feat1, N1, v1f, v1h);

    // sem2
    float* s2af = (float*)(ws + OFF_S2AF);
    _Float16* s2ah = (_Float16*)(ws + OFF_S2AH);
    run_bottleneck(stream, ws, v1f, v1h, 30, 30, 18, 2, 128, 32, 256,
                   F(54), F(55), F(56), F(57), F(58), F(59), F(60), F(61), F(62),
                   F(63), F(64), F(65), s2af, s2ah);
    float* sem2f = (float*)(ws + OFF_S2BF);
    run_bottleneck(stream, ws, s2af, s2ah, 15, 15, 9, 1, 256, 64, 256,
                   F(66), F(67), F(68), F(69), F(70), F(71), F(72), F(73), F(74),
                   nullptr, nullptr, nullptr, sem2f, nullptr);

    // decoder: up1 = relu(bn(deconv(sem2))) + sem1
    float* mean = (float*)(ws + OFF_MEAN);
    float* var  = (float*)(ws + OFF_VAR);
    float* d1  = (float*)(ws + BG_D1);
    float* up1 = (float*)(ws + BG_UP1);
    float* d2  = (float*)(ws + BG_D2);
    deconv3k<<<gr1(128L * N1), 256, 0, stream>>>(sem2f, 256, 15, 15, 9, F(75), F(76), 128, d1);
    chan_reduce<<<128, 256, 0, stream>>>(d1, N1, N1, 1, mean, var);
    bn_apply<<<dim3(gr1(N1), 128), 256, 0, stream>>>(d1, N1, mean, var, F(77), F(78),
                                                     1, 1 /*resid after relu*/, sem1f, up1, nullptr);
    // up2 = relu(bn(deconv(up1))) + trilinear(up1)
    deconv3k<<<gr1(128L * N0), 256, 0, stream>>>(up1, 128, 30, 30, 18, F(79), F(80), 128, d2);
    chan_reduce<<<128, 256, 0, stream>>>(d2, N0, N0, 1, mean, var);
    float* up2 = (float*)(ws + OFF_FEAT);
    bn_apply<<<dim3(gr1(N0), 128), 256, 0, stream>>>(d2, N0, mean, var, F(81), F(82),
                                                     1, 0, nullptr, up2, nullptr);
    trilinear_add<<<gr1(128L * N0), 256, 0, stream>>>(up1, 128, 30, 30, 18, up2, 60, 60, 36);

    // classifier head
    conv_out_k<<<gr1(12L * N0), 256, 0, stream>>>(up2, F(83), F(84), (float*)d_out, N0);
}
